// SelfAttnHAT_87376814670115
// MI455X (gfx1250) — matrix operand order repaired and checked by reading, not compiled, not run on hardware
//
#include <hip/hip_runtime.h>
#include <hip/hip_bf16.h>
#include <math.h>

// ---------------------------------------------------------------------------
// HAT block (FasterViT-style) for MI455X / gfx1250, wave32 + WMMA bf16.
// GEMM-shaped math through v_wmma_f32_16x16x32_bf16; weight panels staged
// into LDS with the Tensor Data Mover (tensor_load_to_lds / s_wait_tensorcnt);
// WMMA fragments fetched as 128-bit vector loads.
// ---------------------------------------------------------------------------

typedef __attribute__((ext_vector_type(16))) __bf16 v16bf;
typedef __attribute__((ext_vector_type(8)))  __bf16 v8bf;
typedef __attribute__((ext_vector_type(8)))  float  v8f;
typedef __attribute__((ext_vector_type(4)))  unsigned int u32x4;
typedef __attribute__((ext_vector_type(8)))  int    i32x8;
typedef __attribute__((ext_vector_type(4)))  int    i32x4;

#define CDIM   128
#define HEADS  4
#define DHEAD  32
#define HIDDEN 512

__device__ __forceinline__ float gelu_exact(float x) {
    return 0.5f * x * (1.0f + erff(x * 0.70710678118654752f));
}

__device__ __forceinline__ v8f wmma_bf16(v16bf a, v16bf b, v8f c) {
    return __builtin_amdgcn_wmma_f32_16x16x32_bf16(false, a, false, b,
                                                   (short)0, c, false, false);
}

__device__ __forceinline__ v16bf combine16(v8bf lo, v8bf hi) {
    return __builtin_shufflevector(lo, hi, 0, 1, 2, 3, 4, 5, 6, 7,
                                           8, 9, 10, 11, 12, 13, 14, 15);
}

// A-fragment (16x32 bf16): per lane-half the K run is contiguous,
// k = half*16 + e (matches B). Load as 2x16B.
__device__ __forceinline__ v16bf load_a_frag(const __bf16* rowptr, int half) {
    const __bf16* p = rowptr + half * 16;
    v8bf lo = *(const v8bf*)(p);
    v8bf hi = *(const v8bf*)(p + 8);
    return combine16(lo, hi);
}

// ------------------------- Tensor Data Mover staging -----------------------
// Stage a [16 x K] bf16 panel (rows contiguous, row stride = K elements) from
// global memory into LDS. 2D tile descriptor per CDNA5 ISA ch.8 (D# groups).
__device__ __forceinline__ void tdm_load_panel(const __bf16* gsrc, unsigned int lds_off,
                                               int K) {
    unsigned long long ga = (unsigned long long)(const void*)gsrc;
    unsigned int td0 = (unsigned int)K;          // tensor_dim0 (elements)
    unsigned int td1 = 16u;                      // tensor_dim1
    unsigned long long st0 = (unsigned long long)K;        // dim0 stride
    unsigned long long st1 = (unsigned long long)K * 16u;  // dim1 stride (unused for 2D)

    u32x4 g0;
    g0[0] = 1u;                                   // count=1, user descriptor
    g0[1] = lds_off;                              // LDS byte address
    g0[2] = (unsigned int)(ga & 0xFFFFFFFFull);   // global_addr[31:0]
    g0[3] = (unsigned int)((ga >> 32) & 0x1FFFFFFull) | (2u << 30);  // addr[56:32], type=2

    i32x8 g1;
    g1[0] = (int)(1u << 16);                      // workgroup_mask=0, data_size=1 (2B)
    g1[1] = (int)((td0 & 0xFFFFu) << 16);         // bits63:48 = tensor_dim0[15:0]
    g1[2] = (int)(((td0 >> 16) & 0xFFFFu) | ((td1 & 0xFFFFu) << 16));
    g1[3] = (int)(((td1 >> 16) & 0xFFFFu) | ((td0 & 0xFFFFu) << 16)); // tile_dim0 = K
    g1[4] = (int)(16u);                           // tile_dim1 = 16, tile_dim2 = 0
    g1[5] = (int)(st0 & 0xFFFFFFFFull);
    g1[6] = (int)(((st0 >> 32) & 0xFFFFu) | ((st1 & 0xFFFFu) << 16));
    g1[7] = (int)((st1 >> 16) & 0xFFFFFFFFull);

    i32x4 gz  = {0, 0, 0, 0};                     // groups 2/3: dims 3/4 unused
    i32x8 gz8 = {0, 0, 0, 0, 0, 0, 0, 0};
    __builtin_amdgcn_tensor_load_to_lds(g0, g1, gz, gz, gz8, 0);
}

// --------------------------- tiny precompute kernels -----------------------

__global__ void posemb_kernel(const float* __restrict__ w1, const float* __restrict__ b1,
                              const float* __restrict__ w2, float* __restrict__ pe, int s) {
    int sidx = blockIdx.x;
    int c    = threadIdx.x;
    int i = sidx / s, j = sidx % s;
    float half = (float)(s / 2);
    float t0 = ((float)i - half) / half;
    float t1 = ((float)j - half) / half;
    float acc = 0.f;
    for (int h = 0; h < HIDDEN; ++h) {
        float hid = fmaxf(t0 * w1[h] + t1 * w1[HIDDEN + h] + b1[h], 0.f);
        acc += hid * w2[h * CDIM + c];
    }
    pe[(size_t)sidx * CDIM + c] = acc;
}

__global__ void cpb_table_kernel(const float* __restrict__ w1, const float* __restrict__ b1,
                                 const float* __restrict__ w2, float* __restrict__ tbl, int ws) {
    int r = blockIdx.x;
    int head = threadIdx.x;
    if (head >= HEADS) return;
    int s = 2 * ws - 1;
    float h0 = (float)(r / s - (ws - 1));
    float w0 = (float)(r % s - (ws - 1));
    h0 = h0 / (float)(ws - 1) * 8.f;
    w0 = w0 / (float)(ws - 1) * 8.f;
    float inv_l8 = 1.f / log2f(8.f);
    float sg0 = (h0 > 0.f) ? 1.f : ((h0 < 0.f) ? -1.f : 0.f);
    float sg1 = (w0 > 0.f) ? 1.f : ((w0 < 0.f) ? -1.f : 0.f);
    float t0 = sg0 * log2f(fabsf(h0) + 1.f) * inv_l8;
    float t1 = sg1 * log2f(fabsf(w0) + 1.f) * inv_l8;
    float acc = 0.f;
    for (int h = 0; h < HIDDEN; ++h) {
        float hid = fmaxf(t0 * w1[h] + t1 * w1[HIDDEN + h] + b1[h], 0.f);
        acc += hid * w2[h * HEADS + head];
    }
    tbl[r * HEADS + head] = acc;
}

__global__ void bias_fill_kernel(const float* __restrict__ tbl, float* __restrict__ bias,
                                 int ws, int N) {
    int idx = blockIdx.x * blockDim.x + threadIdx.x;
    int total = HEADS * N * N;
    if (idx >= total) return;
    int h   = idx / (N * N);
    int rem = idx % (N * N);
    int i = rem / N, j = rem % N;
    int ng = N - ws * ws;
    float v = 0.f;
    if (i >= ng && j >= ng) {
        int a = i - ng, b = j - ng;
        int cia = a / ws, cja = a % ws, cib = b / ws, cjb = b % ws;
        int ridx = (cia - cib + ws - 1) * (2 * ws - 1) + (cja - cjb + ws - 1);
        float t = tbl[ridx * HEADS + h];
        v = 16.f / (1.f + expf(-t));
    }
    bias[idx] = v;
}

__global__ void add_pe_kernel(const float* __restrict__ src, const float* __restrict__ pe,
                              float* __restrict__ dst, int n_per_b) {
    size_t row = blockIdx.x;
    int c = threadIdx.x;
    int n = (int)(row % n_per_b);
    dst[row * CDIM + c] = src[row * CDIM + c] + pe[(size_t)n * CDIM + c];
}

// Transpose + convert weights: W (K x N, f32 row-major) -> WT (N x K, bf16)
__global__ void wt_bf16_kernel(const float* __restrict__ W, __bf16* __restrict__ WT,
                               int K, int N) {
    int idx = blockIdx.x * blockDim.x + threadIdx.x;
    if (idx >= N * K) return;
    int n = idx / K, k = idx % K;
    WT[idx] = (__bf16)W[(size_t)k * N + n];
}

// --------------------------- LayerNorm (f32 -> bf16) -----------------------

__global__ void ln_bf16_kernel(const float* __restrict__ X, const float* __restrict__ g,
                               const float* __restrict__ b, __bf16* __restrict__ Y) {
    __shared__ float red[CDIM];
    size_t row = blockIdx.x;
    int c = threadIdx.x;
    float v = X[row * CDIM + c];
    red[c] = v;
    __syncthreads();
    for (int s2 = CDIM / 2; s2 > 0; s2 >>= 1) { if (c < s2) red[c] += red[c + s2]; __syncthreads(); }
    float mean = red[0] * (1.f / CDIM);
    __syncthreads();
    float d = v - mean;
    red[c] = d * d;
    __syncthreads();
    for (int s2 = CDIM / 2; s2 > 0; s2 >>= 1) { if (c < s2) red[c] += red[c + s2]; __syncthreads(); }
    float var = red[0] * (1.f / CDIM);
    float y = d * rsqrtf(var + 1e-5f) * g[c] + b[c];
    Y[row * CDIM + c] = (__bf16)y;
}

// --------------------------- WMMA GEMM -------------------------------------
// WT is bf16 column-major [N][K]. Per block: TDM-stage the 16-column weight
// panel (16 x K bf16) into LDS, s_wait_tensorcnt, then WMMA over K.
// mode0: store bf16 | mode1: gelu(.+bias) bf16 | mode2: OutF += gamma*(.+bias)
__global__ void gemm_wmma_kernel(const __bf16* __restrict__ A, const __bf16* __restrict__ WT,
                                 const float* __restrict__ bias, const float* __restrict__ gamma,
                                 __bf16* __restrict__ OutB, float* __restrict__ OutF,
                                 int M, int K, int N, int mode) {
    __shared__ __bf16 wtile[16 * 512];
    int lane = threadIdx.x;
    int row0 = blockIdx.x * 16;
    int n0   = blockIdx.y * 16;
    int hl = lane >> 4;
    int ml = lane & 15;

    // async TDM load of weight panel; generic->LDS offset = low 32 bits
    unsigned int lds_off = (unsigned int)(size_t)(void*)wtile;
    tdm_load_panel(WT + (size_t)n0 * K, lds_off, K);

    int m = row0 + ml; if (m >= M) m = M - 1;
    int n = n0 + ml;

    __builtin_amdgcn_s_wait_tensorcnt(0);

    v8f acc = {};
    for (int k0 = 0; k0 < K; k0 += 32) {
        v16bf a = load_a_frag(A + (size_t)m * K + k0, hl);
        // B[k][n]: k = half*16+e contiguous -> one 32B LDS read
        v16bf b = *(const v16bf*)(wtile + ml * K + k0 + hl * 16);
        acc = wmma_bf16(a, b, acc);
    }
    float bv = bias ? bias[n] : 0.f;
    #pragma unroll
    for (int r = 0; r < 8; ++r) {
        int mr = row0 + r + 8 * hl;                 // D row = r + 8*half
        if (mr >= M) continue;
        float v = acc[r] + bv;
        if (mode == 1) v = gelu_exact(v);
        if (mode == 2) OutF[(size_t)mr * N + n] += gamma[n] * v;
        else           OutB[(size_t)mr * N + n] = (__bf16)v;
    }
}

// --------------------------- WMMA attention --------------------------------
__global__ void attn_wmma_kernel(const __bf16* __restrict__ qkv, const float* __restrict__ bias,
                                 __bf16* __restrict__ out, int N, int Npad, float scale) {
    __shared__ float  S[16 * 256];
    __shared__ __bf16 P[16 * 256];
    int lane = threadIdx.x;
    int i0 = blockIdx.x * 16;
    int h  = blockIdx.y;
    int bb = blockIdx.z;
    int hl = lane >> 4;
    int ml = lane & 15;
    size_t base = (size_t)bb * N * (3 * CDIM);

    int qm = i0 + ml; if (qm >= N) qm = N - 1;
    v16bf aq = load_a_frag(qkv + base + (size_t)qm * (3 * CDIM) + h * DHEAD, hl);

    for (int jt = 0; jt < Npad / 16; ++jt) {
        int n = jt * 16 + ml;
        int nc = (n < N) ? n : (N - 1);
        // B[k][n] = K[token n][k], k = half*16+e -> contiguous 32B
        v16bf bk = *(const v16bf*)(qkv + base + (size_t)nc * (3 * CDIM) + CDIM
                                   + h * DHEAD + hl * 16);
        v8f s = {};
        s = wmma_bf16(aq, bk, s);
        #pragma unroll
        for (int r = 0; r < 8; ++r) {
            int mloc = r + 8 * hl;
            int i = i0 + mloc; int ic = (i < N) ? i : (N - 1);
            float v = (n < N) ? (s[r] * scale + bias[((size_t)h * N + ic) * N + n]) : -1e30f;
            S[mloc * Npad + n] = v;
        }
    }
    __syncthreads();
    if (lane < 16) {                                 // softmax, one row per lane
        float mx = -1e30f;
        for (int j2 = 0; j2 < Npad; ++j2) mx = fmaxf(mx, S[lane * Npad + j2]);
        float sum = 0.f;
        for (int j2 = 0; j2 < Npad; ++j2) {
            float e = expf(S[lane * Npad + j2] - mx);
            S[lane * Npad + j2] = e; sum += e;
        }
        float inv = 1.f / sum;
        for (int j2 = 0; j2 < Npad; ++j2) P[lane * Npad + j2] = (__bf16)(S[lane * Npad + j2] * inv);
    }
    __syncthreads();
    v8f o0 = {}, o1 = {};
    for (int kt = 0; kt < Npad; kt += 32) {
        v16bf ap = load_a_frag((const __bf16*)P + ml * Npad + kt, hl);
        v16bf bv0, bv1;
        #pragma unroll
        for (int e = 0; e < 16; ++e) {
            int vr = kt + hl * 16 + e; if (vr >= N) vr = N - 1;   // P col is 0 there
            const __bf16* vrow = qkv + base + (size_t)vr * (3 * CDIM) + 2 * CDIM + h * DHEAD;
            bv0[e] = vrow[ml];
            bv1[e] = vrow[16 + ml];
        }
        o0 = wmma_bf16(ap, bv0, o0);
        o1 = wmma_bf16(ap, bv1, o1);
    }
    #pragma unroll
    for (int r = 0; r < 8; ++r) {
        int i = i0 + r + 8 * hl;
        if (i >= N) continue;
        size_t orow = ((size_t)bb * N + i) * CDIM + h * DHEAD;
        out[orow + ml]      = (__bf16)o0[r];
        out[orow + 16 + ml] = (__bf16)o1[r];
    }
}

// --------------------------- glue kernels ----------------------------------

__global__ void assemble_kernel(const float* __restrict__ x, const float* __restrict__ pe_x,
                                const float* __restrict__ ct, float* __restrict__ xcat) {
    size_t row = blockIdx.x;
    int c = threadIdx.x;
    size_t bx = row / 65;
    int j = (int)(row % 65);
    float v;
    if (j < 16) v = ct[(bx * 16 + j) * CDIM + c];
    else {
        int t = j - 16;
        v = x[(bx * 49 + t) * CDIM + c] + pe_x[(size_t)t * CDIM + c];
    }
    xcat[row * CDIM + c] = v;
}

__global__ void finalize_kernel(const float* __restrict__ xcat, const float* __restrict__ gamma1,
                                float* __restrict__ out) {
    size_t row = blockIdx.x;
    int c = threadIdx.x;
    const size_t XROWS = 200704;                     // 4096*49
    if (row < XROWS) {
        size_t b = row / 49;
        int t = (int)(row % 49);
        int wi = t / 7, wj = t % 7;
        int j = ((wi * 4) / 7) * 4 + ((wj * 4) / 7);
        float up = xcat[(b * 65 + j) * CDIM + c];
        float xv = xcat[(b * 65 + 16 + t) * CDIM + c];
        out[row * CDIM + c] = xv + gamma1[c] * up;
    } else {
        size_t r2 = row - XROWS;                     // 0..65535 -> ct_out
        size_t bx = r2 / 16;
        int j = (int)(r2 % 16);
        out[XROWS * CDIM + r2 * CDIM + c] = xcat[(bx * 65 + j) * CDIM + c];
    }
}

// --------------------------- host launcher ---------------------------------

extern "C" void kernel_launch(void* const* d_in, const int* in_sizes, int n_in,
                              void* d_out, int out_size, void* d_ws, size_t ws_size,
                              hipStream_t stream) {
    (void)in_sizes; (void)n_in; (void)out_size; (void)ws_size;
    const float* x           = (const float*)d_in[0];
    const float* ct_in       = (const float*)d_in[1];
    const float* pe_w1       = (const float*)d_in[2];
    const float* pe_b1       = (const float*)d_in[3];
    const float* pe_w2       = (const float*)d_in[4];
    const float* hpe_w1      = (const float*)d_in[5];
    const float* hpe_b1      = (const float*)d_in[6];
    const float* hpe_w2      = (const float*)d_in[7];
    const float* n1_g        = (const float*)d_in[8];
    const float* n1_b        = (const float*)d_in[9];
    const float* n2_g        = (const float*)d_in[10];
    const float* n2_b        = (const float*)d_in[11];
    const float* hn1_g       = (const float*)d_in[12];
    const float* hn1_b       = (const float*)d_in[13];
    const float* hn2_g       = (const float*)d_in[14];
    const float* hn2_b       = (const float*)d_in[15];
    const float* attn_qkv_w  = (const float*)d_in[16];
    const float* attn_proj_w = (const float*)d_in[17];
    const float* attn_proj_b = (const float*)d_in[18];
    const float* attn_cpb_w1 = (const float*)d_in[19];
    const float* attn_cpb_b1 = (const float*)d_in[20];
    const float* attn_cpb_w2 = (const float*)d_in[21];
    const float* hat_qkv_w   = (const float*)d_in[22];
    const float* hat_proj_w  = (const float*)d_in[23];
    const float* hat_proj_b  = (const float*)d_in[24];
    const float* hat_cpb_w1  = (const float*)d_in[25];
    const float* hat_cpb_b1  = (const float*)d_in[26];
    const float* hat_cpb_w2  = (const float*)d_in[27];
    const float* mlp_w1      = (const float*)d_in[28];
    const float* mlp_b1      = (const float*)d_in[29];
    const float* mlp_w2      = (const float*)d_in[30];
    const float* mlp_b2      = (const float*)d_in[31];
    const float* hmlp_w1     = (const float*)d_in[32];
    const float* hmlp_b1     = (const float*)d_in[33];
    const float* hmlp_w2     = (const float*)d_in[34];
    const float* hmlp_b2     = (const float*)d_in[35];
    const float* gamma1      = (const float*)d_in[36];
    const float* gamma2      = (const float*)d_in[37];
    const float* gamma3      = (const float*)d_in[38];
    const float* gamma4      = (const float*)d_in[39];
    float* out = (float*)d_out;

    char* wp = (char*)d_ws;
    auto carve = [&](size_t bytes) -> void* {
        void* p = (void*)wp;
        wp += (bytes + 255) & ~(size_t)255;
        return p;
    };
    float*  pe_x    = (float*)carve((size_t)49 * 128 * 4);
    float*  pe_ct   = (float*)carve((size_t)256 * 128 * 4);
    float*  tbl7    = (float*)carve((size_t)169 * 4 * 4);
    float*  tbl16   = (float*)carve((size_t)961 * 4 * 4);
    float*  bias65  = (float*)carve((size_t)4 * 65 * 65 * 4);
    float*  bias256 = (float*)carve((size_t)4 * 256 * 256 * 4);
    // bf16 column-major weights [N][K]
    __bf16* qkvT    = (__bf16*)carve((size_t)384 * 128 * 2);
    __bf16* projT   = (__bf16*)carve((size_t)128 * 128 * 2);
    __bf16* mlp1T   = (__bf16*)carve((size_t)512 * 128 * 2);
    __bf16* mlp2T   = (__bf16*)carve((size_t)128 * 512 * 2);
    __bf16* hqkvT   = (__bf16*)carve((size_t)384 * 128 * 2);
    __bf16* hprojT  = (__bf16*)carve((size_t)128 * 128 * 2);
    __bf16* hmlp1T  = (__bf16*)carve((size_t)512 * 128 * 2);
    __bf16* hmlp2T  = (__bf16*)carve((size_t)128 * 512 * 2);
    float*  ct_buf  = (float*)carve((size_t)65536 * 128 * 4);
    float*  xcat    = (float*)carve((size_t)266240 * 128 * 4);
    __bf16* lnbuf   = (__bf16*)carve((size_t)266240 * 128 * 2);
    __bf16* attnout = (__bf16*)carve((size_t)266240 * 128 * 2);
    __bf16* big     = (__bf16*)carve((size_t)266240 * 512 * 2); // qkv / MLP-hidden (shared)

    const float scale = 0.17677669529663689f;  // 32^-0.5

    // ---- precompute ----
    posemb_kernel<<<49, 128, 0, stream>>>(pe_w1, pe_b1, pe_w2, pe_x, 7);
    posemb_kernel<<<256, 128, 0, stream>>>(hpe_w1, hpe_b1, hpe_w2, pe_ct, 16);
    cpb_table_kernel<<<169, 32, 0, stream>>>(attn_cpb_w1, attn_cpb_b1, attn_cpb_w2, tbl7, 7);
    cpb_table_kernel<<<961, 32, 0, stream>>>(hat_cpb_w1, hat_cpb_b1, hat_cpb_w2, tbl16, 16);
    bias_fill_kernel<<<(4 * 65 * 65 + 255) / 256, 256, 0, stream>>>(tbl7, bias65, 7, 65);
    bias_fill_kernel<<<(4 * 256 * 256 + 255) / 256, 256, 0, stream>>>(tbl16, bias256, 16, 256);
    wt_bf16_kernel<<<(384 * 128 + 255) / 256, 256, 0, stream>>>(attn_qkv_w, qkvT, 128, 384);
    wt_bf16_kernel<<<(128 * 128 + 255) / 256, 256, 0, stream>>>(attn_proj_w, projT, 128, 128);
    wt_bf16_kernel<<<(512 * 128 + 255) / 256, 256, 0, stream>>>(mlp_w1, mlp1T, 128, 512);
    wt_bf16_kernel<<<(512 * 128 + 255) / 256, 256, 0, stream>>>(mlp_w2, mlp2T, 512, 128);
    wt_bf16_kernel<<<(384 * 128 + 255) / 256, 256, 0, stream>>>(hat_qkv_w, hqkvT, 128, 384);
    wt_bf16_kernel<<<(128 * 128 + 255) / 256, 256, 0, stream>>>(hat_proj_w, hprojT, 128, 128);
    wt_bf16_kernel<<<(512 * 128 + 255) / 256, 256, 0, stream>>>(hmlp_w1, hmlp1T, 128, 512);
    wt_bf16_kernel<<<(512 * 128 + 255) / 256, 256, 0, stream>>>(hmlp_w2, hmlp2T, 512, 128);
    add_pe_kernel<<<65536, 128, 0, stream>>>(ct_in, pe_ct, ct_buf, 256);

    // ---- HAT (carrier) branch: attn + MLP, residual in ct_buf ----
    ln_bf16_kernel<<<65536, 128, 0, stream>>>(ct_buf, hn1_g, hn1_b, lnbuf);
    gemm_wmma_kernel<<<dim3(65536 / 16, 384 / 16), 32, 0, stream>>>(
        lnbuf, hqkvT, nullptr, nullptr, big, nullptr, 65536, 128, 384, 0);
    attn_wmma_kernel<<<dim3(16, HEADS, 256), 32, 0, stream>>>(big, bias256, attnout, 256, 256, scale);
    gemm_wmma_kernel<<<dim3(65536 / 16, 128 / 16), 32, 0, stream>>>(
        attnout, hprojT, hat_proj_b, gamma1, nullptr, ct_buf, 65536, 128, 128, 2);
    ln_bf16_kernel<<<65536, 128, 0, stream>>>(ct_buf, hn2_g, hn2_b, lnbuf);
    gemm_wmma_kernel<<<dim3(65536 / 16, 512 / 16), 32, 0, stream>>>(
        lnbuf, hmlp1T, hmlp_b1, nullptr, big, nullptr, 65536, 128, 512, 1);
    gemm_wmma_kernel<<<dim3(65536 / 16, 128 / 16), 32, 0, stream>>>(
        big, hmlp2T, hmlp_b2, gamma2, nullptr, ct_buf, 65536, 512, 128, 2);

    // ---- concat [carrier, window+pe] ----
    assemble_kernel<<<266240, 128, 0, stream>>>(x, pe_x, ct_buf, xcat);

    // ---- main branch: attn (N=65, pad 96) + MLP, residual in xcat ----
    ln_bf16_kernel<<<266240, 128, 0, stream>>>(xcat, n1_g, n1_b, lnbuf);
    gemm_wmma_kernel<<<dim3(266240 / 16, 384 / 16), 32, 0, stream>>>(
        lnbuf, qkvT, nullptr, nullptr, big, nullptr, 266240, 128, 384, 0);
    attn_wmma_kernel<<<dim3(5, HEADS, 4096), 32, 0, stream>>>(big, bias65, attnout, 65, 96, scale);
    gemm_wmma_kernel<<<dim3(266240 / 16, 128 / 16), 32, 0, stream>>>(
        attnout, projT, attn_proj_b, gamma3, nullptr, xcat, 266240, 128, 128, 2);
    ln_bf16_kernel<<<266240, 128, 0, stream>>>(xcat, n2_g, n2_b, lnbuf);
    gemm_wmma_kernel<<<dim3(266240 / 16, 512 / 16), 32, 0, stream>>>(
        lnbuf, mlp1T, mlp_b1, nullptr, big, nullptr, 266240, 128, 512, 1);
    gemm_wmma_kernel<<<dim3(266240 / 16, 128 / 16), 32, 0, stream>>>(
        big, mlp2T, mlp_b2, gamma4, nullptr, xcat, 266240, 512, 128, 2);

    // ---- split, upsample carrier into windows, emit (x, ct_out) ----
    finalize_kernel<<<266240, 128, 0, stream>>>(xcat, gamma1, out);
}
